// Model_11484742550244
// MI455X (gfx1250) — matrix operand order repaired and checked by reading, not compiled, not run on hardware
//
#include <hip/hip_runtime.h>
#include <hip/hip_bf16.h>

// ---------- types ----------
typedef __attribute__((ext_vector_type(8)))  unsigned short u16x8;
typedef __attribute__((ext_vector_type(16))) unsigned short u16x16;
typedef __attribute__((ext_vector_type(16))) __bf16        v16bf;
typedef __attribute__((ext_vector_type(8)))  float         v8f;

#define SHUF16(lo, hi) __builtin_shufflevector((lo), (hi), 0,1,2,3,4,5,6,7,8,9,10,11,12,13,14,15)
#define WMMA_BF16(a, b, c) __builtin_amdgcn_wmma_f32_16x16x32_bf16( \
    false, __builtin_bit_cast(v16bf, (a)), false, __builtin_bit_cast(v16bf, (b)), \
    (short)0, (c), false, false)

// async global->LDS copy (CDNA5 GLOBAL_LOAD_ASYNC_TO_LDS_B128, GVS mode).
// ldsaddr: per-lane LDS byte address (low 32 bits of generic pointer);
// voff: per-lane 32-bit byte offset; sbase: uniform 64-bit global base.
#define ASYNC_LDS_B128(ldsaddr, voff, sbase) \
  asm volatile("global_load_async_to_lds_b128 %0, %1, %2" \
               :: "v"(ldsaddr), "v"(voff), "s"(sbase) : "memory")
#define WAIT_ASYNC() asm volatile("s_wait_asynccnt 0x0" ::: "memory")

static __device__ inline unsigned short f2bf(float f) {
  unsigned u = __builtin_bit_cast(unsigned, f);
  u += 0x7FFFu + ((u >> 16) & 1u);           // round-to-nearest-even
  return (unsigned short)(u >> 16);
}
static __device__ inline float sigm(float x) { return 1.0f / (1.0f + __expf(-x)); }
static __device__ inline unsigned ldsAddr(const void* p) {
  return (unsigned)(unsigned long long)p;    // aperture: addr[31:0] == LDS byte address
}

// ---------- model dims ----------
#define H1 241
#define W1 80
#define H2 105
#define W2 36
#define K2 9216           // 32*32*9
#define MX 6720           // 64*105
#define KX 1152
#define NX 384

// ================= f32 -> bf16 convert =================
__global__ void k_cvt(const float* __restrict__ src, unsigned short* __restrict__ dst, int n) {
  int i = blockIdx.x * 256 + threadIdx.x;
  if (i < n) dst[i] = f2bf(src[i]);
}

// ================= conv1 (normalize + conv + relu, bf16 out) =================
__global__ void __launch_bounds__(256) k_conv1(const float* __restrict__ x,
                                               const float* __restrict__ mean,
                                               const float* __restrict__ stdv,
                                               const float* __restrict__ w,
                                               const float* __restrict__ bias,
                                               unsigned short* __restrict__ a1) {
  int idx = blockIdx.x * 256 + threadIdx.x;          // exactly 64*32*241*80
  int wpos = idx % W1;
  int t = idx / W1;
  int h = t % H1; t /= H1;
  int co = t % 32; int n = t / 32;
  int iw0 = 2 * wpos, iw1 = iw0 + 1;
  float m0 = mean[iw0], m1 = mean[iw1];
  float r0 = 1.0f / stdv[iw0], r1 = 1.0f / stdv[iw1];
  const float* xr = x + (size_t)n * 512 * 161;
  const float* wr = w + co * 64;                     // (co,0,kh,kw)
  float acc = bias[co];
#pragma unroll
  for (int kh = 0; kh < 32; ++kh) {
    const float* xi = xr + (size_t)(2 * h + kh) * 161;
    acc += (xi[iw0] - m0) * r0 * wr[kh * 2 + 0];
    acc += (xi[iw1] - m1) * r1 * wr[kh * 2 + 1];
  }
  a1[idx] = f2bf(fmaxf(acc, 0.0f));
}

// ================= conv2 implicit GEMM (WMMA bf16) =================
// D(M=pos, N=co) = sum_k A_im2col(M,k) * W(k,co); conv2_w natural layout == [co][k].
// Per-k im2col offsets are position-independent -> precomputed once into LDS.
__global__ void __launch_bounds__(256) k_conv2(const unsigned short* __restrict__ a1,
                                               const unsigned short* __restrict__ w2b,
                                               const float* __restrict__ c2b,
                                               unsigned short* __restrict__ x2) {
  __shared__ unsigned short As[128][40];             // 10,240 B (80B padded rows)
  __shared__ int kofs[K2];                           // 36,864 B offset table
  const int tid = threadIdx.x;
  const int lane = tid & 31, wave = tid >> 5;
  const int mbase = blockIdx.x * 128;
  for (int k = tid; k < K2; k += 256) {              // one-time decode (divs hoisted here)
    int ci = k / 288; int rem = k - ci * 288;
    int kh = rem / 9; int kw = rem - kh * 9;
    kofs[k] = ci * (H1 * W1) + kh * W1 + kw;
  }
  // gather coords (fixed per thread)
  const int mloc = tid >> 1;
  const int khalf = (tid & 1) << 4;
  int mg = mbase + mloc;
  int fw = mg % W2; int t0 = mg / W2; int th = t0 % H2; int nb = t0 / H2;
  const unsigned short* athr = a1 + (size_t)nb * 32 * H1 * W1 + (2 * th) * W1 + 2 * fw;
  // wmma coords
  const int arow = (wave << 4) + (lane & 15);
  const int ak0 = (lane >> 4) << 4;
  const int bcol = lane & 15;
  const int bk0 = (lane >> 4) << 4;
  v8f acc0 = {}, acc1 = {};
  __syncthreads();                                   // kofs ready
  for (int kc = 0; kc < 288; ++kc) {
    const int kbase = kc << 5;
#pragma unroll
    for (int i = 0; i < 16; i += 4) {
      int4 o = *(const int4*)&kofs[kbase + khalf + i];
      As[mloc][khalf + i + 0] = athr[o.x];
      As[mloc][khalf + i + 1] = athr[o.y];
      As[mloc][khalf + i + 2] = athr[o.z];
      As[mloc][khalf + i + 3] = athr[o.w];
    }
    __syncthreads();
    u16x16 af = SHUF16(*(const u16x8*)&As[arow][ak0], *(const u16x8*)&As[arow][ak0 + 8]);
    const unsigned short* bp0 = w2b + (size_t)bcol * K2 + kbase + bk0;
    const unsigned short* bp1 = bp0 + (size_t)16 * K2;
    u16x16 b0 = SHUF16(*(const u16x8*)bp0, *(const u16x8*)(bp0 + 8));
    u16x16 b1 = SHUF16(*(const u16x8*)bp1, *(const u16x8*)(bp1 + 8));
    acc0 = WMMA_BF16(af, b0, acc0);
    acc1 = WMMA_BF16(af, b1, acc1);
    __syncthreads();
  }
  // epilogue: incremental position decode (no divisions in the loop)
  const int mrh = (lane >> 4) << 3;
  int m2 = mbase + (wave << 4) + mrh;
  int fw2 = m2 % W2; int t2 = m2 / W2; int th2 = t2 % H2; int n2 = t2 / H2;
#pragma unroll
  for (int r = 0; r < 8; ++r) {
    size_t rowo = ((size_t)n2 * H2 + th2) * KX;      // x (b,t, c*36+fw)
    float v0 = fmaxf(acc0[r] + c2b[bcol], 0.0f);
    float v1 = fmaxf(acc1[r] + c2b[16 + bcol], 0.0f);
    x2[rowo + (size_t)bcol * W2 + fw2] = f2bf(v0);
    x2[rowo + (size_t)(16 + bcol) * W2 + fw2] = f2bf(v1);
    if (++fw2 == W2) { fw2 = 0; if (++th2 == H2) { th2 = 0; ++n2; } }
  }
}

// ================= generic bf16 GEMM: C[M][N] = A[M][K] * B[N][K]^T + bias =================
// Tiles staged with async global->LDS (ASYNCcnt) instead of VGPR round trips.
__global__ void __launch_bounds__(256) k_gemm(const unsigned short* __restrict__ A,
                                              const unsigned short* __restrict__ B,
                                              const float* __restrict__ bias,
                                              float* __restrict__ C,
                                              int M, int N, int K) {
  __shared__ unsigned short As[64][40];
  __shared__ unsigned short Bs[64][40];
  const int tid = threadIdx.x, lane = tid & 31, wave = tid >> 5;
  const int mbase = blockIdx.x * 64, nbase = blockIdx.y * 64;
  const int lr = tid >> 2;                 // 0..63
  const int lk = (tid & 3) << 3;           // 0,8,16,24
  const int mt = wave & 3, ntb = wave >> 2;
  const int arow = (mt << 4) + (lane & 15);
  const int ak0 = (lane >> 4) << 4;
  const int bcol = lane & 15;
  const int bk0 = (lane >> 4) << 4;
  const unsigned ldsa = ldsAddr(&As[lr][lk]);
  const unsigned ldsb = ldsAddr(&Bs[lr][lk]);
  const unsigned long long ga = (unsigned long long)A;
  const unsigned long long gb = (unsigned long long)B;
  const unsigned aoff = (unsigned)(((mbase + lr) * K + lk) * 2);
  const unsigned boff = (unsigned)(((nbase + lr) * K + lk) * 2);
  v8f acc0 = {}, acc1 = {};
  for (int kb = 0; kb < K; kb += 32) {
    ASYNC_LDS_B128(ldsa, aoff + (unsigned)(kb << 1), ga);
    ASYNC_LDS_B128(ldsb, boff + (unsigned)(kb << 1), gb);
    WAIT_ASYNC();
    __syncthreads();
    u16x16 af = SHUF16(*(const u16x8*)&As[arow][ak0], *(const u16x8*)&As[arow][ak0 + 8]);
    int br0 = (ntb << 4) + bcol;
    int br1 = br0 + 32;
    u16x16 b0 = SHUF16(*(const u16x8*)&Bs[br0][bk0], *(const u16x8*)&Bs[br0][bk0 + 8]);
    u16x16 b1 = SHUF16(*(const u16x8*)&Bs[br1][bk0], *(const u16x8*)&Bs[br1][bk0 + 8]);
    acc0 = WMMA_BF16(af, b0, acc0);
    acc1 = WMMA_BF16(af, b1, acc1);
    __syncthreads();
  }
  const int mrh = (lane >> 4) << 3;
#pragma unroll
  for (int r = 0; r < 8; ++r) {
    int m = mbase + (mt << 4) + mrh + r;
    int j0 = nbase + (ntb << 4) + bcol;
    int j1 = j0 + 32;
    C[(size_t)m * N + j0] = acc0[r] + bias[j0];
    C[(size_t)m * N + j1] = acc1[r] + bias[j1];
  }
}

// ================= encoder GRU scan (single WG, WMMA per step) =================
__global__ void __launch_bounds__(256) k_enc_gru(const float* __restrict__ xp,
                                                 const unsigned short* __restrict__ whhb,
                                                 const float* __restrict__ bhh,
                                                 float* __restrict__ gh,
                                                 float* __restrict__ eh) {
  __shared__ unsigned short hbf[64][136];            // bf16 h, padded rows (272B)
  __shared__ float hf[64][128];                      // f32 h
  const int tid = threadIdx.x, lane = tid & 31, wave = tid >> 5;
  for (int i = tid; i < 64 * 128; i += 256) {
    hf[i >> 7][i & 127] = 0.0f;
    hbf[i >> 7][i & 127] = 0;
  }
  __syncthreads();
  const int mt = wave & 3, ng = wave >> 2;
  const int arow = (mt << 4) + (lane & 15);
  const int ak0 = (lane >> 4) << 4;
  const int bcol = lane & 15;
  const int bk0 = (lane >> 4) << 4;
  const int mrh = (lane >> 4) << 3;
  const v8f vzero = {};
  for (int t = 0; t < H2; ++t) {
    // ---- gh = h @ w_hh^T  (M=64, N=384, K=128) ----
    v8f acc[12];
#pragma unroll
    for (int i = 0; i < 12; ++i) acc[i] = vzero;
#pragma unroll
    for (int kcc = 0; kcc < 4; ++kcc) {
      int kb = kcc << 5;
      u16x16 af = SHUF16(*(const u16x8*)&hbf[arow][kb + ak0],
                         *(const u16x8*)&hbf[arow][kb + ak0 + 8]);
#pragma unroll
      for (int i = 0; i < 12; ++i) {
        int nt = ng + (i << 1);
        const unsigned short* bp = whhb + (size_t)(nt * 16 + bcol) * 128 + kb + bk0;
        u16x16 bf = SHUF16(*(const u16x8*)bp, *(const u16x8*)(bp + 8));
        acc[i] = WMMA_BF16(af, bf, acc[i]);
      }
    }
#pragma unroll
    for (int i = 0; i < 12; ++i) {
      int nt = ng + (i << 1);
#pragma unroll
      for (int r = 0; r < 8; ++r) {
        int m = (mt << 4) + mrh + r;
        gh[m * NX + nt * 16 + bcol] = acc[i][r];
      }
    }
    __threadfence_block();
    __syncthreads();
    // ---- gates ----
    for (int i = tid; i < 64 * 128; i += 256) {
      int m = i >> 7, d = i & 127;
      const float* xprow = xp + ((size_t)m * H2 + t) * NX;
      float hr = gh[m * NX + d] + bhh[d];
      float hz = gh[m * NX + 128 + d] + bhh[128 + d];
      float hn = gh[m * NX + 256 + d] + bhh[256 + d];
      float r = sigm(xprow[d] + hr);
      float z = sigm(xprow[128 + d] + hz);
      float nn = tanhf(xprow[256 + d] + r * hn);
      float hnew = (1.0f - z) * nn + z * hf[m][d];
      hf[m][d] = hnew;
      hbf[m][d] = f2bf(hnew);
      eh[((size_t)m * H2 + t) * 128 + d] = hnew;
    }
    __syncthreads();
  }
}

// ================= decoder input projection =================
__global__ void k_ip(const int* __restrict__ y, const float* __restrict__ emb,
                     const float* __restrict__ dwih, const float* __restrict__ dbih,
                     float* __restrict__ ip) {
  int idx = blockIdx.x * 256 + threadIdx.x;          // exactly 4032*384
  int j = idx % NX; int m = idx / NX;
  int s = m % 63; int b = m / 63;
  int tok = y[b * 64 + s];
  const float* e = emb + tok * 32;
  const float* w = dwih + j * 32;
  float acc = dbih[j];
#pragma unroll
  for (int k = 0; k < 32; ++k) acc += e[k] * w[k];
  ip[idx] = acc;
}

// ================= decoder scan + attention + FC (one WG per batch row) =================
__global__ void __launch_bounds__(256) k_dec(const float* __restrict__ eh,
                                             const float* __restrict__ ip,
                                             const float* __restrict__ dwhh,
                                             const float* __restrict__ dbhh,
                                             const float* __restrict__ fcw,
                                             const float* __restrict__ fcb,
                                             float* __restrict__ out) {
  __shared__ float ehs[H2 * 128];
  __shared__ float ghs[NX];
  __shared__ float hx[128], hx2[128], ctx[128];
  __shared__ float satt[H2];
  __shared__ float red[2];
  const int tid = threadIdx.x;
  const int b = blockIdx.x;
  // async prefetch of this batch row's encoder states into LDS
  const unsigned long long ge = (unsigned long long)(eh + (size_t)b * H2 * 128);
  for (int i4 = tid; i4 < (H2 * 128) / 4; i4 += 256) {
    ASYNC_LDS_B128(ldsAddr(&ehs[i4 * 4]), (unsigned)(i4 * 16), ge);
  }
  if (tid < 128) hx[tid] = 0.0f;
  WAIT_ASYNC();
  __syncthreads();
  for (int s = 0; s < 63; ++s) {
    for (int j = tid; j < NX; j += 256) {            // gh = dec_w_hh @ hx + b_hh
      const float* w = dwhh + (size_t)j * 128;
      float a = dbhh[j];
      for (int k = 0; k < 128; ++k) a += w[k] * hx[k];
      ghs[j] = a;
    }
    __syncthreads();
    const float* iprow = ip + ((size_t)b * 63 + s) * NX;
    if (tid < 128) {
      int d = tid;
      float r = sigm(iprow[d] + ghs[d]);
      float z = sigm(iprow[128 + d] + ghs[128 + d]);
      float nn = tanhf(iprow[256 + d] + r * ghs[256 + d]);
      hx2[d] = (1.0f - z) * nn + z * hx[d];
    }
    __syncthreads();
    if (tid < H2) {                                  // attention scores
      const float* er = &ehs[tid * 128];
      float a = 0.0f;
      for (int d = 0; d < 128; ++d) a += er[d] * hx2[d];
      satt[tid] = a;
    }
    __syncthreads();
    if (tid == 0) {
      float mx = satt[0];
      for (int t2 = 1; t2 < H2; ++t2) mx = fmaxf(mx, satt[t2]);
      red[0] = mx;
    }
    __syncthreads();
    if (tid < H2) satt[tid] = __expf(satt[tid] - red[0]);
    __syncthreads();
    if (tid == 0) {
      float sm = 0.0f;
      for (int t2 = 0; t2 < H2; ++t2) sm += satt[t2];
      red[1] = 1.0f / sm;
    }
    __syncthreads();
    if (tid < 128) {                                 // ctx; carry = ctx (reference quirk)
      int d = tid;
      float c = 0.0f;
      for (int t2 = 0; t2 < H2; ++t2) c += satt[t2] * ehs[t2 * 128 + d];
      c *= red[1];
      ctx[d] = c;
      hx[d] = c;
    }
    __syncthreads();
    if (tid < 32) {                                  // fused FC
      const float* w = fcw + tid * 128;
      float o = fcb[tid];
      for (int d = 0; d < 128; ++d) o += w[d] * ctx[d];
      out[((size_t)b * 63 + s) * 32 + tid] = o;
    }
    __syncthreads();
  }
}

// ================= host =================
extern "C" void kernel_launch(void* const* d_in, const int* in_sizes, int n_in,
                              void* d_out, int out_size, void* d_ws, size_t ws_size,
                              hipStream_t stream) {
  (void)in_sizes; (void)n_in; (void)out_size; (void)ws_size;
  const float* x      = (const float*)d_in[0];
  const int*   y      = (const int*)  d_in[1];
  const float* mean   = (const float*)d_in[2];
  const float* stdv   = (const float*)d_in[3];
  const float* conv1w = (const float*)d_in[4];
  const float* conv1b = (const float*)d_in[5];
  const float* conv2w = (const float*)d_in[6];
  const float* conv2b = (const float*)d_in[7];
  const float* wih    = (const float*)d_in[8];
  const float* whh    = (const float*)d_in[9];
  const float* bih    = (const float*)d_in[10];
  const float* bhh    = (const float*)d_in[11];
  const float* emb    = (const float*)d_in[12];
  const float* dwih   = (const float*)d_in[13];
  const float* dwhh   = (const float*)d_in[14];
  const float* dbih   = (const float*)d_in[15];
  const float* dbhh   = (const float*)d_in[16];
  const float* fcw    = (const float*)d_in[17];
  const float* fcb    = (const float*)d_in[18];
  float* out = (float*)d_out;

  char* base = (char*)d_ws;
  size_t off = 0;
  auto take = [&](size_t bytes) -> char* {
    off = (off + 255) & ~(size_t)255;
    char* p = base + off;
    off += bytes;
    return p;
  };
  unsigned short* a1   = (unsigned short*)take((size_t)64 * 32 * H1 * W1 * 2); // conv1 out bf16
  unsigned short* w2b  = (unsigned short*)take((size_t)32 * K2 * 2);           // conv2 w bf16
  unsigned short* x2   = (unsigned short*)take((size_t)MX * KX * 2);           // conv2 out bf16
  unsigned short* wihb = (unsigned short*)take((size_t)NX * KX * 2);           // gru_w_ih bf16
  unsigned short* whhb = (unsigned short*)take((size_t)NX * 128 * 2);          // gru_w_hh bf16
  float* xp = (float*)take((size_t)MX * NX * 4);                               // input proj
  float* eh = (float*)take((size_t)64 * H2 * 128 * 4);                         // enc states
  float* gh = (float*)take((size_t)64 * NX * 4);                               // GRU scratch
  float* ip = (float*)take((size_t)64 * 63 * NX * 4);                          // dec input proj

  k_cvt<<<(32 * K2 + 255) / 256, 256, 0, stream>>>(conv2w, w2b, 32 * K2);
  k_cvt<<<(NX * KX + 255) / 256, 256, 0, stream>>>(wih, wihb, NX * KX);
  k_cvt<<<(NX * 128 + 255) / 256, 256, 0, stream>>>(whh, whhb, NX * 128);
  k_conv1<<<154240, 256, 0, stream>>>(x, mean, stdv, conv1w, conv1b, a1);
  k_conv2<<<1890, 256, 0, stream>>>(a1, w2b, conv2b, x2);
  k_gemm<<<dim3(MX / 64, NX / 64), 256, 0, stream>>>(x2, wihb, bih, xp, MX, NX, KX);
  k_enc_gru<<<1, 256, 0, stream>>>(xp, whhb, bhh, gh, eh);
  k_ip<<<6048, 256, 0, stream>>>(y, emb, dwih, dbih, ip);
  k_dec<<<64, 256, 0, stream>>>(eh, ip, dwhh, dbhh, fcw, fcb, out);
}
